// SelfAttention_14559939133818
// MI455X (gfx1250) — matrix operand order repaired and checked by reading, not compiled, not run on hardware
//
#include <hip/hip_runtime.h>

#define D_MODEL 1024
#define N_HEADS 16
#define HEAD_DIM 64
#define SEQ_T 2048
#define BATCH 4

typedef __bf16 bf16;
typedef __attribute__((ext_vector_type(16))) __bf16 v16bf;
typedef __attribute__((ext_vector_type(8)))  __bf16 v8bf;
typedef __attribute__((ext_vector_type(8)))  float  v8f;

// ---------------------------------------------------------------------------
// WMMA fragment helpers (contiguous k = 16*half + s order for A and B, wave32)
// ---------------------------------------------------------------------------

// A fragment (16x32 bf16, row-major source, ld = row stride in elements).
// lane: m = lane&15, half = lane>>4. element s holds K = 16*half + s
// -> one contiguous 16-element (32B) run at +16*half.
static __device__ __forceinline__ v16bf load_a_frag(const bf16* __restrict__ base,
                                                    int ld, int lane) {
  const int m = lane & 15, half = lane >> 4;
  const bf16* p = base + (size_t)m * ld + 16 * half;
  return *reinterpret_cast<const v16bf*>(p);
}

// B fragment (32x16 bf16) where B[k][n] = srcT[n][k] (i.e. source stored
// N-major with row stride ldt). lane: n = lane&15, half = lane>>4.
// VGPR v holds K = 16*half + 2v, 2v+1 -> one contiguous 16-element (32B) load.
static __device__ __forceinline__ v16bf load_bT_frag(const bf16* __restrict__ baseT,
                                                     int ldt, int lane) {
  const int n = lane & 15, half = lane >> 4;
  return *reinterpret_cast<const v16bf*>(baseT + (size_t)n * ldt + 16 * half);
}

static __device__ __forceinline__ v8f wmma_bf16(v16bf a, v16bf b, v8f c) {
  return __builtin_amdgcn_wmma_f32_16x16x32_bf16(false, a, false, b,
                                                 (short)0, c, false, false);
}

// 64x64 register-blocked GEMM core: 4 A-frags x 4 B-frags -> 16 WMMAs / k-step
// (32 FLOP per byte of fragment traffic; accumulators live in 128 VGPRs).
static __device__ __forceinline__ void gemm_64x64(const bf16* __restrict__ Abase,
                                                  const bf16* __restrict__ Bbase,
                                                  int lane, v8f acc[16]) {
  for (int k = 0; k < D_MODEL; k += 32) {
    // Speculative prefetch 4 k-steps ahead. Lanes fan across all 64 tile rows
    // so each global_prefetch_b8 touches 32 cachelines; past-end prefetches
    // are TH=RT-speculative and silently dropped (ISA 10.5) -> no guard.
    __builtin_prefetch(Abase + (size_t)lane * D_MODEL + k + 128);
    __builtin_prefetch(Abase + (size_t)(lane + 32) * D_MODEL + k + 128);
    __builtin_prefetch(Bbase + (size_t)lane * D_MODEL + k + 128);
    __builtin_prefetch(Bbase + (size_t)(lane + 32) * D_MODEL + k + 128);

    v16bf a[4], b[4];
#pragma unroll
    for (int i = 0; i < 4; ++i)
      a[i] = load_a_frag(Abase + (size_t)i * 16 * D_MODEL + k, D_MODEL, lane);
#pragma unroll
    for (int j = 0; j < 4; ++j)
      b[j] = load_bT_frag(Bbase + (size_t)j * 16 * D_MODEL + k, D_MODEL, lane);
#pragma unroll
    for (int i = 0; i < 4; ++i)
#pragma unroll
      for (int j = 0; j < 4; ++j)
        acc[i * 4 + j] = wmma_bf16(a[i], b[j], acc[i * 4 + j]);
  }
}

// ---------------------------------------------------------------------------
// Precision conversion / weight transposition
// ---------------------------------------------------------------------------
__global__ void cvt_f32_to_bf16(const float* __restrict__ in,
                                bf16* __restrict__ out, int n) {
  int i = blockIdx.x * blockDim.x + threadIdx.x;
  if (i < n) out[i] = (bf16)in[i];
}

// out[n*rows + k] = in[k*cols + n]   (rows = K dim, cols = N dim)
__global__ void transpose_to_bf16(const float* __restrict__ in,
                                  bf16* __restrict__ out, int rows, int cols) {
  int i = blockIdx.x * blockDim.x + threadIdx.x;
  if (i < rows * cols) {
    int n = i / rows, k = i % rows;
    out[i] = (bf16)in[(size_t)k * cols + n];
  }
}

// ---------------------------------------------------------------------------
// Kernel 1: qkv = x @ W_qkv + b_qkv, scattered head-major.
// Q,K -> [B,H,T,64] bf16 ; V -> transposed [B,H,64,T] bf16 (for PV fragment).
// One wave per 64x64 macro tile. 64-wide N tiles align with head boundaries
// and 64-row M tiles lie inside one batch, so section/head/batch selection is
// wave-uniform scalar work.
// ---------------------------------------------------------------------------
__global__ void __launch_bounds__(128)
qkv_gemm_kernel(const bf16* __restrict__ Xb,     // [8192][1024]
                const bf16* __restrict__ WqkvT,  // [3072][1024] (N-major)
                const float* __restrict__ bias,  // [3072]
                bf16* __restrict__ Qw, bf16* __restrict__ Kw,
                bf16* __restrict__ Vt) {
  const int lane = threadIdx.x & 31;
  const int wv = threadIdx.x >> 5;
  const int tilesN = (3 * D_MODEL) / 64;  // 48
  const int w = blockIdx.x * 4 + wv;
  const int tm = w / tilesN, tn = w % tilesN;

  const bf16* Abase = Xb + (size_t)(tm * 64) * D_MODEL;
  const bf16* Bbase = WqkvT + (size_t)(tn * 64) * D_MODEL;

  v8f acc[16];
#pragma unroll
  for (int i = 0; i < 16; ++i) acc[i] = (v8f){};
  gemm_64x64(Abase, Bbase, lane, acc);

  const int n = lane & 15, half = lane >> 4;
  // Wave-uniform decomposition of the output tile position.
  const int sec = tn >> 4;                 // 0=Q, 1=K, 2=V
  const int hh = tn & 15;                  // head
  const int bb = (tm * 64) >> 11;          // batch (tile fits in one batch)
  const int hbase = bb * N_HEADS + hh;
  const int tbase = tm * 64 - bb * SEQ_T;  // token offset within batch

  if (sec < 2) {
    bf16* base = (sec == 0 ? Qw : Kw) + (size_t)hbase * SEQ_T * HEAD_DIM;
#pragma unroll
    for (int j = 0; j < 4; ++j) {
      const float bv = bias[tn * 64 + j * 16 + n];
      const int dh = j * 16 + n;
#pragma unroll
      for (int i = 0; i < 4; ++i)
#pragma unroll
        for (int v = 0; v < 8; ++v) {
          const int t = tbase + i * 16 + v + 8 * half;
          base[(size_t)t * HEAD_DIM + dh] = (bf16)(acc[i * 4 + j][v] + bv);
        }
    }
  } else {
    bf16* base = Vt + (size_t)hbase * HEAD_DIM * SEQ_T;
#pragma unroll
    for (int j = 0; j < 4; ++j) {
      const float bv = bias[tn * 64 + j * 16 + n];
      const int dh = j * 16 + n;
#pragma unroll
      for (int i = 0; i < 4; ++i)
#pragma unroll
        for (int v = 0; v < 8; ++v) {
          const int t = tbase + i * 16 + v + 8 * half;
          base[(size_t)dh * SEQ_T + t] = (bf16)(acc[i * 4 + j][v] + bv);
        }
    }
  }
}

// ---------------------------------------------------------------------------
// Kernel 2: causal flash attention. One wave per (b, h, 16-row q tile).
// Online softmax; P reshaped D-layout -> A-layout through a 16x32 LDS tile.
// ---------------------------------------------------------------------------
__global__ void __launch_bounds__(128)
attn_kernel(const bf16* __restrict__ Qw, const bf16* __restrict__ Kw,
            const bf16* __restrict__ Vt, bf16* __restrict__ Yw) {
  __shared__ alignas(32) bf16 plds[4][16 * 32];
  const int lane = threadIdx.x & 31;
  const int wv = threadIdx.x >> 5;
  bf16* P = &plds[wv][0];

  const int QT = SEQ_T / 16;  // 128 q tiles
  const int w = blockIdx.x * 4 + wv;
  const int qt = w % QT;
  const int bh = w / QT;  // b*16 + h

  const bf16* Qb = Qw + ((size_t)bh * SEQ_T + qt * 16) * HEAD_DIM;
  const bf16* Kb = Kw + (size_t)bh * SEQ_T * HEAD_DIM;
  const bf16* Vb = Vt + (size_t)bh * HEAD_DIM * SEQ_T;
  const int n = lane & 15, half = lane >> 4;

  // Q fragments for the whole tile (Dh = 64 -> two 16x32 chunks), kept resident.
  v16bf aQ0 = load_a_frag(Qb, HEAD_DIM, lane);
  v16bf aQ1 = load_a_frag(Qb + 32, HEAD_DIM, lane);

  v8f acc0 = {}, acc1 = {}, acc2 = {}, acc3 = {};
  float rmax[8], rsum[8];
#pragma unroll
  for (int v = 0; v < 8; ++v) { rmax[v] = -3.0e38f; rsum[v] = 0.f; }

  const int nkt = qt + 1;      // causal: kv tiles 0..qt
  const float scale = 0.125f;  // 1/sqrt(64)

  for (int t0 = 0; t0 < nkt; t0 += 2) {
    const int t1 = t0 + 1;
    const bool has1 = (t1 < nkt);

    v8f s0 = {};
    {
      const bf16* KT = Kb + (size_t)t0 * 16 * HEAD_DIM;
      s0 = wmma_bf16(aQ0, load_bT_frag(KT, HEAD_DIM, lane), s0);
      s0 = wmma_bf16(aQ1, load_bT_frag(KT + 32, HEAD_DIM, lane), s0);
    }
    v8f s1 = {};
    if (has1) {
      const bf16* KT = Kb + (size_t)t1 * 16 * HEAD_DIM;
      s1 = wmma_bf16(aQ0, load_bT_frag(KT, HEAD_DIM, lane), s1);
      s1 = wmma_bf16(aQ1, load_bT_frag(KT + 32, HEAD_DIM, lane), s1);
    }

    // mask + online softmax update; row stats reduced across the 16-lane half
    float p0[8], p1[8], alpha[8];
#pragma unroll
    for (int v = 0; v < 8; ++v) {
      const int qi = qt * 16 + v + 8 * half;
      float x0 = s0[v] * scale, x1 = s1[v] * scale;
      if (t0 * 16 + n > qi) x0 = -3.0e38f;
      if (!has1 || (t1 * 16 + n > qi)) x1 = -3.0e38f;
      float mv = fmaxf(x0, x1);
      mv = fmaxf(mv, __shfl_xor(mv, 1, 32));
      mv = fmaxf(mv, __shfl_xor(mv, 2, 32));
      mv = fmaxf(mv, __shfl_xor(mv, 4, 32));
      mv = fmaxf(mv, __shfl_xor(mv, 8, 32));
      float newmax = fmaxf(rmax[v], mv);
      float a = __expf(rmax[v] - newmax);
      float e0 = __expf(x0 - newmax);
      float e1 = __expf(x1 - newmax);
      float rs = e0 + e1;
      rs += __shfl_xor(rs, 1, 32);
      rs += __shfl_xor(rs, 2, 32);
      rs += __shfl_xor(rs, 4, 32);
      rs += __shfl_xor(rs, 8, 32);
      rsum[v] = a * rsum[v] + rs;
      rmax[v] = newmax;
      alpha[v] = a;
      p0[v] = e0;
      p1[v] = e1;
    }

#pragma unroll
    for (int v = 0; v < 8; ++v) {
      acc0[v] *= alpha[v]; acc1[v] *= alpha[v];
      acc2[v] *= alpha[v]; acc3[v] *= alpha[v];
    }

    // Stage exp(P) (16x32) into LDS in D-layout, read back in A-layout.
#pragma unroll
    for (int v = 0; v < 8; ++v) {
      const int m = v + 8 * half;
      P[m * 32 + n] = (bf16)p0[v];
      P[m * 32 + 16 + n] = (bf16)p1[v];
    }
    asm volatile("s_wait_dscnt 0" ::: "memory");
    v16bf aP = load_a_frag(P, 32, lane);

    // PV: B[k][d] = V[t0*16+k][d] = Vt[d][t0*16+k]  (contiguous per lane)
    const bf16* VT = Vb + t0 * 16;
    acc0 = wmma_bf16(aP, load_bT_frag(VT, SEQ_T, lane), acc0);
    acc1 = wmma_bf16(aP, load_bT_frag(VT + (size_t)16 * SEQ_T, SEQ_T, lane), acc1);
    acc2 = wmma_bf16(aP, load_bT_frag(VT + (size_t)32 * SEQ_T, SEQ_T, lane), acc2);
    acc3 = wmma_bf16(aP, load_bT_frag(VT + (size_t)48 * SEQ_T, SEQ_T, lane), acc3);
  }

  const int b = bh >> 4, h = bh & 15;
#pragma unroll
  for (int v = 0; v < 8; ++v) {
    const float inv = 1.0f / rsum[v];
    const size_t row =
        ((size_t)b * SEQ_T + qt * 16 + v + 8 * half) * D_MODEL + h * HEAD_DIM + n;
    Yw[row + 0]  = (bf16)(acc0[v] * inv);
    Yw[row + 16] = (bf16)(acc1[v] * inv);
    Yw[row + 32] = (bf16)(acc2[v] * inv);
    Yw[row + 48] = (bf16)(acc3[v] * inv);
  }
}

// ---------------------------------------------------------------------------
// Kernel 3: out = y @ W_proj + b_proj (fp32 output), 64x64 per-wave tiles.
// ---------------------------------------------------------------------------
__global__ void __launch_bounds__(128)
proj_gemm_kernel(const bf16* __restrict__ Yb, const bf16* __restrict__ WprojT,
                 const float* __restrict__ bias, float* __restrict__ out) {
  const int lane = threadIdx.x & 31;
  const int wv = threadIdx.x >> 5;
  const int tilesN = D_MODEL / 64;  // 16
  const int w = blockIdx.x * 4 + wv;
  const int tm = w / tilesN, tn = w % tilesN;

  const bf16* Abase = Yb + (size_t)(tm * 64) * D_MODEL;
  const bf16* Bbase = WprojT + (size_t)(tn * 64) * D_MODEL;

  v8f acc[16];
#pragma unroll
  for (int i = 0; i < 16; ++i) acc[i] = (v8f){};
  gemm_64x64(Abase, Bbase, lane, acc);

  const int n = lane & 15, half = lane >> 4;
#pragma unroll
  for (int j = 0; j < 4; ++j) {
    const int c = tn * 64 + j * 16 + n;
    const float bv = bias[c];
#pragma unroll
    for (int i = 0; i < 4; ++i)
#pragma unroll
      for (int v = 0; v < 8; ++v)
        out[((size_t)(tm * 64 + i * 16 + v + 8 * half)) * D_MODEL + c] =
            acc[i * 4 + j][v] + bv;
  }
}

// ---------------------------------------------------------------------------
extern "C" void kernel_launch(void* const* d_in, const int* in_sizes, int n_in,
                              void* d_out, int out_size, void* d_ws, size_t ws_size,
                              hipStream_t stream) {
  (void)in_sizes; (void)n_in; (void)out_size; (void)ws_size;
  const float* x = (const float*)d_in[0];
  const float* W_qkv = (const float*)d_in[1];
  const float* b_qkv = (const float*)d_in[2];
  const float* W_proj = (const float*)d_in[3];
  const float* b_proj = (const float*)d_in[4];
  float* out = (float*)d_out;

  const size_t elems = (size_t)BATCH * SEQ_T * D_MODEL;  // 8388608
  bf16* Qw = (bf16*)d_ws;                 // [B,H,T,64]
  bf16* Kw = Qw + elems;                  // [B,H,T,64]
  bf16* Vt = Kw + elems;                  // [B,H,64,T]
  bf16* Yw = Vt + elems;                  // [B,T,C]
  bf16* Xb = Yw + elems;                  // [B*T, C]
  bf16* WqkvT = Xb + elems;               // [3C, C]
  bf16* WprojT = WqkvT + (size_t)3 * D_MODEL * D_MODEL;  // [C, C]

  cvt_f32_to_bf16<<<(int)((elems + 255) / 256), 256, 0, stream>>>(x, Xb, (int)elems);
  transpose_to_bf16<<<(3 * D_MODEL * D_MODEL + 255) / 256, 256, 0, stream>>>(
      W_qkv, WqkvT, D_MODEL, 3 * D_MODEL);
  transpose_to_bf16<<<(D_MODEL * D_MODEL + 255) / 256, 256, 0, stream>>>(
      W_proj, WprojT, D_MODEL, D_MODEL);

  // QKV: 128 x 48 macro tiles of 64x64, 4 waves per 128-thread block
  qkv_gemm_kernel<<<128 * 48 / 4, 128, 0, stream>>>(Xb, WqkvT, b_qkv, Qw, Kw, Vt);
  // Attention: 4*16 head-batches x 128 q tiles
  attn_kernel<<<BATCH * N_HEADS * (SEQ_T / 16) / 4, 128, 0, stream>>>(Qw, Kw, Vt, Yw);
  // Proj: 128 x 16 macro tiles of 64x64
  proj_gemm_kernel<<<128 * 16 / 4, 128, 0, stream>>>(Yw, WprojT, b_proj, out);
}
